// MultiHeadedAttention_v2_22960895164973
// MI455X (gfx1250) — compile-verified, matrix operand order checked by reading, not run on hardware
//
#include <hip/hip_runtime.h>
#include <hip/hip_bf16.h>

#define HD   12
#define DM   768
#define DH   64
#define LSEQ 2048
#define NB   4

typedef __attribute__((ext_vector_type(16))) __bf16 bf16x16;
typedef __attribute__((ext_vector_type(2)))  __bf16 bf16x2;
typedef __attribute__((ext_vector_type(8)))  float  f32x8;

union Frag16 { bf16x16 v; unsigned int p[8]; };

__device__ __forceinline__ unsigned int pk2(float lo, float hi) {
  bf16x2 t = {(__bf16)lo, (__bf16)hi};          // -> v_cvt_pk_bf16_f32
  return __builtin_bit_cast(unsigned int, t);
}
__device__ __forceinline__ unsigned short bf16b(float f) {
  return __builtin_bit_cast(unsigned short, (__bf16)f);
}

// Load one 16-bit 16x32 A/B fragment for this lane from a bf16 row at `base`
// (base already points at the fragment's K-origin). Two 16-byte loads.
__device__ __forceinline__ void load_frag(Frag16& f, const unsigned short* base, int kh) {
  uint4 lo = *(const uint4*)(base + (kh << 3));
  uint4 hi = *(const uint4*)(base + 16 + (kh << 3));
  f.p[0] = lo.x; f.p[1] = lo.y; f.p[2] = lo.z; f.p[3] = lo.w;
  f.p[4] = hi.x; f.p[5] = hi.y; f.p[6] = hi.z; f.p[7] = hi.w;
}

// ---------------- f32 -> bf16 elementwise conversion (8 elems/thread) --------------
__global__ __launch_bounds__(256)
void cvt_kernel(const float* __restrict__ in, unsigned short* __restrict__ out, int n) {
  int i = (blockIdx.x * 256 + threadIdx.x) * 8;
  if (i + 8 <= n) {
    float4 a = *(const float4*)(in + i);
    float4 b = *(const float4*)(in + i + 4);
    uint4 o = {pk2(a.x, a.y), pk2(a.z, a.w), pk2(b.x, b.y), pk2(b.z, b.w)};
    *(uint4*)(out + i) = o;
  }
}

// ---------------- int mask -> bitmask (1 bit per element, 1 word per wave) ---------
__global__ __launch_bounds__(256)
void maskpack_kernel(const int* __restrict__ mask, unsigned int* __restrict__ mp) {
  int w = blockIdx.x * 8 + (threadIdx.x >> 5);
  int lane = threadIdx.x & 31;
  unsigned long long bal = __ballot(mask[(long long)w * 32 + lane] != 0);
  if (lane == 0) mp[w] = (unsigned int)bal;
}

// ---------------- Projection GEMM: out = X(bf16) @ W^T(bf16) + bias, stored bf16 ---
__global__ __launch_bounds__(256)
void proj_kernel(const unsigned short* __restrict__ X, const unsigned short* __restrict__ W,
                 const float* __restrict__ bias, unsigned short* __restrict__ out,
                 int vtrans) {
  const int lane = threadIdx.x & 31;
  const int wave = threadIdx.x >> 5;
  const int kh = lane >> 4, ln = lane & 15;
  const int rbase = blockIdx.x * 128 + (wave & 3) * 32;
  const int cbase = blockIdx.y * 64  + (wave >> 2) * 32;

  const unsigned short* xr0 = X + (rbase + ln) * DM;
  const unsigned short* xr1 = X + (rbase + 16 + ln) * DM;
  const unsigned short* wr0 = W + (cbase + ln) * DM;
  const unsigned short* wr1 = W + (cbase + 16 + ln) * DM;

  f32x8 acc[2][2] = {};
#pragma unroll 2
  for (int kb = 0; kb < DM; kb += 32) {
    Frag16 a[2], b[2];
    load_frag(a[0], xr0 + kb, kh);
    load_frag(a[1], xr1 + kb, kh);
    load_frag(b[0], wr0 + kb, kh);
    load_frag(b[1], wr1 + kb, kh);
#pragma unroll
    for (int mt = 0; mt < 2; ++mt)
#pragma unroll
      for (int nt = 0; nt < 2; ++nt)
        acc[mt][nt] = __builtin_amdgcn_wmma_f32_16x16x32_bf16(
            false, a[mt].v, false, b[nt].v, (short)0, acc[mt][nt], false, false);
  }
#pragma unroll
  for (int mt = 0; mt < 2; ++mt)
#pragma unroll
    for (int nt = 0; nt < 2; ++nt) {
      int col = cbase + nt * 16 + ln;
      int h = col >> 6, d = col & 63;
      float bv = bias[col];
#pragma unroll
      for (int v = 0; v < 8; ++v) {
        int row = rbase + mt * 16 + v + (kh << 3);
        int bidx = row >> 11, lrow = row & (LSEQ - 1);
        int idx = vtrans ? (((bidx * HD + h) * DH + d) * LSEQ + lrow)
                         : (((bidx * HD + h) * LSEQ + lrow) * DH + d);
        out[idx] = bf16b(acc[mt][nt][v] + bv);
      }
    }
}

// ---------------- Flash attention: S^T trick, no LDS, no barriers ------------------
__global__ __launch_bounds__(128)
void attn_kernel(const unsigned short* __restrict__ qbuf,
                 const unsigned short* __restrict__ kbuf,
                 const unsigned short* __restrict__ vT,
                 const unsigned int* __restrict__ mpack,
                 unsigned short* __restrict__ xout) {
  const int lane = threadIdx.x & 31;
  const int wave = threadIdx.x >> 5;
  const int kh = lane >> 4, ln = lane & 15;
  const int bh = blockIdx.x;
  const int b = bh / HD, h = bh - b * HD;
  const int qbase = blockIdx.y * 64 + wave * 16;

  // Q fragments for both halves of D=64 (B operand of S^T), held for whole kernel
  Frag16 qf[2];
  {
    const unsigned short* qr = qbuf + (bh * LSEQ + qbase + ln) * DH;
    load_frag(qf[0], qr, kh);
    load_frag(qf[1], qr + 32, kh);
  }
  // per-lane packed mask row for this lane's query
  const unsigned int* mrow = mpack + (long long)(b * LSEQ + qbase + ln) * (LSEQ / 32);

  f32x8 oacc[4] = {};
  float mstat = -1e30f, lstat = 0.f;

  for (int kb = 0; kb < LSEQ; kb += 32) {
    // S^T = K @ Q^T : C-layout lane = query (ln), VGPR = key (vv + 8*kh + 16*nt)
    f32x8 ST[2] = {};
#pragma unroll
    for (int s = 0; s < 2; ++s)
#pragma unroll
      for (int nt = 0; nt < 2; ++nt) {
        Frag16 kf;
        load_frag(kf, kbuf + (bh * LSEQ + kb + nt * 16 + ln) * DH + s * 32, kh);
        ST[nt] = __builtin_amdgcn_wmma_f32_16x16x32_bf16(
            false, kf.v, false, qf[s].v, (short)0, ST[nt], false, false);
      }
    __builtin_prefetch(kbuf + (bh * LSEQ + kb + 32 + ln) * DH, 0, 1);

    // scale + mask (single packed word covers all 32 keys of this step)
    unsigned int mw = mrow[kb >> 5];
    float mx = -1e30f;
#pragma unroll
    for (int nt = 0; nt < 2; ++nt)
#pragma unroll
      for (int vv = 0; vv < 8; ++vv) {
        float sv = ST[nt][vv] * 0.125f;        // 1/sqrt(64)
        if (!((mw >> (nt * 16 + vv + (kh << 3))) & 1)) sv = -1e9f;
        ST[nt][vv] = sv;
        mx = fmaxf(mx, sv);
      }
    // full row stats: in-lane over 16 keys + one half-merge shuffle
    mx = fmaxf(mx, __shfl_xor(mx, 16, 32));
    float nm = fmaxf(mstat, mx);
    float alpha = __expf(mstat - nm);
    float rs = 0.f;
#pragma unroll
    for (int nt = 0; nt < 2; ++nt)
#pragma unroll
      for (int vv = 0; vv < 8; ++vv) {
        float p = __expf(ST[nt][vv] - nm);
        ST[nt][vv] = p;
        rs += p;
      }
    rs += __shfl_xor(rs, 16, 32);
    lstat = lstat * alpha + rs;
    mstat = nm;

    // rescale O rows (alpha lives in lane == its query; broadcast per row)
#pragma unroll
    for (int v = 0; v < 8; ++v) {
      float av = __shfl(alpha, v + (kh << 3), 32);
#pragma unroll
      for (int dt = 0; dt < 4; ++dt) oacc[dt][v] *= av;
    }

    // P A-fragment by pure register repack (no LDS): pair v needs keys
    // (v>>2)*16 + kh*8 + (v&3)*2 .. +1 for query ln == exactly ST[v>>2][(v&3)*2..+1]
    Frag16 pf;
#pragma unroll
    for (int v = 0; v < 8; ++v)
      pf.p[v] = pk2(ST[v >> 2][(v & 3) * 2], ST[v >> 2][(v & 3) * 2 + 1]);

    // O += P @ V  (V stored transposed: key pairs contiguous)
#pragma unroll
    for (int dt = 0; dt < 4; ++dt) {
      Frag16 vf;
      load_frag(vf, vT + (bh * DH + dt * 16 + ln) * LSEQ + kb, kh);
      oacc[dt] = __builtin_amdgcn_wmma_f32_16x16x32_bf16(
          false, pf.v, false, vf.v, (short)0, oacc[dt], false, false);
    }
  }
  // normalize and write attention output (bf16, (N,L,H*D))
  float inv = 1.f / lstat;
#pragma unroll
  for (int v = 0; v < 8; ++v) {
    float iv = __shfl(inv, v + (kh << 3), 32);
    int q = qbase + v + (kh << 3);
#pragma unroll
    for (int dt = 0; dt < 4; ++dt)
      xout[(b * LSEQ + q) * DM + h * DH + dt * 16 + ln] = bf16b(oacc[dt][v] * iv);
  }
}

// ---------------- Final FC: d_out(f32) = x(bf16) @ FC_w^T(bf16) + FC_b --------------
__global__ __launch_bounds__(256)
void fc_kernel(const unsigned short* __restrict__ Xb, const unsigned short* __restrict__ W,
               const float* __restrict__ bias, float* __restrict__ out) {
  const int lane = threadIdx.x & 31;
  const int wave = threadIdx.x >> 5;
  const int kh = lane >> 4, ln = lane & 15;
  const int rbase = blockIdx.x * 128 + (wave & 3) * 32;
  const int cbase = blockIdx.y * 64  + (wave >> 2) * 32;

  const unsigned short* xr0 = Xb + (rbase + ln) * DM;
  const unsigned short* xr1 = Xb + (rbase + 16 + ln) * DM;
  const unsigned short* wr0 = W + (cbase + ln) * DM;
  const unsigned short* wr1 = W + (cbase + 16 + ln) * DM;

  f32x8 acc[2][2] = {};
#pragma unroll 2
  for (int kb = 0; kb < DM; kb += 32) {
    Frag16 a[2], b[2];
    load_frag(a[0], xr0 + kb, kh);
    load_frag(a[1], xr1 + kb, kh);
    load_frag(b[0], wr0 + kb, kh);
    load_frag(b[1], wr1 + kb, kh);
#pragma unroll
    for (int mt = 0; mt < 2; ++mt)
#pragma unroll
      for (int nt = 0; nt < 2; ++nt)
        acc[mt][nt] = __builtin_amdgcn_wmma_f32_16x16x32_bf16(
            false, a[mt].v, false, b[nt].v, (short)0, acc[mt][nt], false, false);
  }
#pragma unroll
  for (int mt = 0; mt < 2; ++mt)
#pragma unroll
    for (int nt = 0; nt < 2; ++nt) {
      int col = cbase + nt * 16 + ln;
      float bv = bias[col];
#pragma unroll
      for (int v = 0; v < 8; ++v) {
        int row = rbase + mt * 16 + v + (kh << 3);
        out[row * DM + col] = acc[mt][nt][v] + bv;
      }
    }
}

extern "C" void kernel_launch(void* const* d_in, const int* in_sizes, int n_in,
                              void* d_out, int out_size, void* d_ws, size_t ws_size,
                              hipStream_t stream) {
  const float* Q    = (const float*)d_in[0];
  const float* K    = (const float*)d_in[1];
  const float* V    = (const float*)d_in[2];
  const int*   mask = (const int*)d_in[3];
  const float* Wq_w = (const float*)d_in[4];
  const float* Wq_b = (const float*)d_in[5];
  const float* Wk_w = (const float*)d_in[6];
  const float* Wk_b = (const float*)d_in[7];
  const float* Wv_w = (const float*)d_in[8];
  const float* Wv_b = (const float*)d_in[9];
  const float* FC_w = (const float*)d_in[10];
  const float* FC_b = (const float*)d_in[11];

  const size_t SZ  = (size_t)NB * HD * LSEQ * DH;  // 6,291,456 bf16 elems
  const size_t WSZ = (size_t)DM * DM;              // 589,824 bf16 elems
  unsigned short* qbuf = (unsigned short*)d_ws;
  unsigned short* kbuf = qbuf + SZ;
  unsigned short* vTb  = qbuf + 2 * SZ;
  unsigned short* tmp  = qbuf + 3 * SZ;            // bf16 activations, later xbuf
  unsigned short* Wqb  = qbuf + 4 * SZ;
  unsigned short* Wkb  = Wqb + WSZ;
  unsigned short* Wvb  = Wqb + 2 * WSZ;
  unsigned short* FCwb = Wqb + 3 * WSZ;
  unsigned int*   mpk  = (unsigned int*)(Wqb + 4 * WSZ);  // 524,288 words (2 MB)

  const int nAct   = (int)SZ;
  const int gAct   = nAct / 2048;                  // 3072 blocks
  const int gW     = (int)WSZ / 2048;              // 288 blocks
  const int nWords = NB * LSEQ * (LSEQ / 32);      // 524,288
  dim3 gg(NB * LSEQ / 128, DM / 64);               // 64 x 12

  // weights -> bf16, mask -> bitmask (once)
  cvt_kernel<<<gW, 256, 0, stream>>>(Wq_w, Wqb, (int)WSZ);
  cvt_kernel<<<gW, 256, 0, stream>>>(Wk_w, Wkb, (int)WSZ);
  cvt_kernel<<<gW, 256, 0, stream>>>(Wv_w, Wvb, (int)WSZ);
  cvt_kernel<<<gW, 256, 0, stream>>>(FC_w, FCwb, (int)WSZ);
  maskpack_kernel<<<nWords / 8, 256, 0, stream>>>(mask, mpk);

  // activations through shared tmp buffer (stream order serializes reuse)
  cvt_kernel<<<gAct, 256, 0, stream>>>(Q, tmp, nAct);
  proj_kernel<<<gg, 256, 0, stream>>>(tmp, Wqb, Wq_b, qbuf, 0);
  cvt_kernel<<<gAct, 256, 0, stream>>>(K, tmp, nAct);
  proj_kernel<<<gg, 256, 0, stream>>>(tmp, Wkb, Wk_b, kbuf, 0);
  cvt_kernel<<<gAct, 256, 0, stream>>>(V, tmp, nAct);
  proj_kernel<<<gg, 256, 0, stream>>>(tmp, Wvb, Wv_b, vTb, 1);

  unsigned short* xbuf = tmp;                      // recycle
  attn_kernel<<<dim3(NB * HD, LSEQ / 64), 128, 0, stream>>>(qbuf, kbuf, vTb, mpk, xbuf);

  fc_kernel<<<gg, 256, 0, stream>>>(xbuf, FCwb, FC_b, (float*)d_out);
}
